// CrossAtten_83502754169464
// MI455X (gfx1250) — compile-verified, matrix operand order checked by reading, not run on hardware
//
#include <hip/hip_runtime.h>
#include <hip/hip_bf16.h>

// MI455X / gfx1250 cross-attention (B=4,H=16,I=J=2048,D=128), bf16 WMMA pipeline.
// ~155 GFLOP vs ~200MB HBM -> compute bound; all matmuls via v_wmma_f32_16x16x32_bf16.
// Attention K/V tiles staged block-cooperatively with global_load_async_to_lds_b128
// (ASYNCcnt / s_wait_asynccnt), double-buffered in LDS (160KB of the 320KB WGP LDS).

typedef __bf16 bf16_t;
typedef __attribute__((ext_vector_type(16))) __bf16 v16bf;
typedef __attribute__((ext_vector_type(8)))  __bf16 v8bf;
typedef __attribute__((ext_vector_type(8)))  float  v8f;

#define DHEAD 128
#define SEQ   2048
#define NBH   64              // B*H
#define NROWS (NBH * SEQ)     // 131072
#define JTILE 128
#define MTILE 128

static __device__ __forceinline__ int lane_id() { return (int)(threadIdx.x & 31); }

// D = A*B + C, bf16 inputs, f32 accumulate (CDNA5 VOP3P, 8-arg form)
static __device__ __forceinline__ v8f wmma_bf16(v16bf a, v16bf b, v8f c) {
  return __builtin_amdgcn_wmma_f32_16x16x32_bf16(false, a, false, b, (short)0, c,
                                                 false, false);
}

// Async copy: 16 bytes per lane, global (saddr base + 32-bit vaddr offset) -> LDS.
// LDS destination byte-offset = low 32 bits of the generic shared pointer.
static __device__ __forceinline__ void async_b128(unsigned lds_off,
                                                  unsigned long long gbase,
                                                  int goff) {
  asm volatile("global_load_async_to_lds_b128 %0, %1, %2"
               :: "v"(lds_off), "v"(goff), "s"(gbase)
               : "memory");
}
static __device__ __forceinline__ void wait_async_0()  {
  asm volatile("s_wait_asynccnt 0x0" ::: "memory");
}
static __device__ __forceinline__ void wait_async_16() {
  asm volatile("s_wait_asynccnt 0x10" ::: "memory");
}

// 16x32 bf16 A/B fragment from a row-major bf16 matrix (leading dim ld, elements).
// CDNA5 layout (ISA 7.12.2): lane l -> row (row0 + (l&15));
// lanes 0-15 hold K chunks {0..7, 16..23}, lanes 16-31 hold {8..15, 24..31}.
static __device__ __forceinline__ v16bf frag_ld_bf16(const bf16_t* p, long ld,
                                                     int row0, long k0) {
  int l = lane_id(); int r = l & 15; int hs = l >> 4;
  const bf16_t* s = p + (long)(row0 + r) * ld + k0 + hs * 8;
  v8bf lo = *(const v8bf*)(s);        // 16B load (global_load_b128 / ds_load_b128)
  v8bf hi = *(const v8bf*)(s + 16);
  v16bf f;
#pragma unroll
  for (int i = 0; i < 8; ++i) { f[i] = lo[i]; f[i + 8] = hi[i]; }
  return f;
}

// Same fragment from an fp32 row-major matrix, converting to bf16 on the fly.
static __device__ __forceinline__ v16bf frag_ld_f32(const float* p, long ld,
                                                    int row0, long k0) {
  int l = lane_id(); int r = l & 15; int hs = l >> 4;
  const float* s = p + (long)(row0 + r) * ld + k0 + hs * 8;
  v16bf f;
#pragma unroll
  for (int i = 0; i < 8; ++i) {
    f[i]     = (bf16_t)s[i];
    f[i + 8] = (bf16_t)s[i + 16];
  }
  return f;
}

// ---------------- Kernel 1: projections --------------------------------------
// Y[row, n] = sum_k X[row,k] * W[n,k]   (torch Linear, W is [ncols, 128])
// cols [0,128)   -> row-major bf16 store to Yrow (scaled)
// cols [128,256) -> transposed bf16 store to YT laid out [bh][d][SEQ]
__global__ void __launch_bounds__(256)
proj_kernel(const float* __restrict__ X, const float* __restrict__ W,
            bf16_t* __restrict__ Yrow, bf16_t* __restrict__ YT,
            int ncols, float scale) {
  int wave = (int)(threadIdx.x >> 5);
  int row0 = ((int)blockIdx.x * 8 + wave) * 16;
  int l = lane_id(); int r = l & 15; int hs = l >> 4;

  v16bf a[4];
#pragma unroll
  for (int kc = 0; kc < 4; ++kc) a[kc] = frag_ld_f32(X, DHEAD, row0, kc * 32);

  for (int nt = 0; nt < ncols / 16; ++nt) {
    v8f acc = {};
#pragma unroll
    for (int kc = 0; kc < 4; ++kc) {
      v16bf b = frag_ld_f32(W, DHEAD, nt * 16, kc * 32);
      acc = wmma_bf16(a[kc], b, acc);
    }
    if (nt * 16 < DHEAD) {
      bf16_t* o = Yrow + (long)(row0 + hs * 8) * DHEAD + nt * 16 + r;
#pragma unroll
      for (int v = 0; v < 8; ++v) o[(long)v * DHEAD] = (bf16_t)(acc[v] * scale);
    } else {
      int bh = row0 / SEQ;
      int j0 = (row0 % SEQ) + hs * 8;
      int d  = nt * 16 - DHEAD + r;
      v8bf pk;
#pragma unroll
      for (int v = 0; v < 8; ++v) pk[v] = (bf16_t)(acc[v] * scale);
      *(v8bf*)(YT + ((long)bh * DHEAD + d) * SEQ + j0) = pk;
    }
  }
}

// ---------------- Kernel 2: flash attention ----------------------------------
// Per block: one (b,h), 128 q-rows (8 waves x 16). Online softmax over j-tiles.
// K/V tiles double-buffered in LDS via async-to-LDS copies.
__global__ void __launch_bounds__(256)
attn_kernel(const bf16_t* __restrict__ QP, const bf16_t* __restrict__ KP,
            const bf16_t* __restrict__ VT, bf16_t* __restrict__ AO) {
  extern __shared__ char smem[];
  bf16_t* Kt  = (bf16_t*)(smem);            // [2][128*128] bf16 = 64 KB
  bf16_t* Vt  = (bf16_t*)(smem + 65536);    // [2][128*128] bf16 = 64 KB
  bf16_t* Pst = (bf16_t*)(smem + 131072);   // [8 waves][16*128] bf16 = 32 KB

  int tid  = (int)threadIdx.x;
  int wave = tid >> 5;
  int l = tid & 31; int r = l & 15; int hs = l >> 4;
  int bh = (int)blockIdx.x >> 4;
  int mt = (int)blockIdx.x & 15;
  int row0 = mt * MTILE + wave * 16;

  const bf16_t* Q = QP + (long)bh * SEQ * DHEAD;   // [i][d], pre-scaled
  const bf16_t* K = KP + (long)bh * SEQ * DHEAD;   // [j][d]
  const bf16_t* V = VT + (long)bh * DHEAD * SEQ;   // [d][j] (transposed)
  unsigned long long Kg = (unsigned long long)K;
  unsigned long long Vg = (unsigned long long)V;
  unsigned ldsK = (unsigned)(size_t)Kt;
  unsigned ldsV = (unsigned)(size_t)Vt;

  // Stage one 128x128 bf16 K tile (linear) + V tile (16 rows/slot) into buffer.
  // 16 async b128 ops per wave per tile (8 K + 8 V), 512B each wave-wide.
  auto stage = [&](int j0, int buf) {
    unsigned lk = ldsK + (unsigned)buf * 32768u + (unsigned)tid * 16u;
    int gk = j0 * (DHEAD * 2) + tid * 16;
#pragma unroll
    for (int s = 0; s < 8; ++s) async_b128(lk + s * 4096u, Kg, gk + s * 4096);
    int d = tid >> 4, c = tid & 15;
    unsigned lv = ldsV + (unsigned)buf * 32768u + (unsigned)(d * 128 + c * 8) * 2u;
    int gv = (d * SEQ + j0 + c * 8) * 2;
#pragma unroll
    for (int s = 0; s < 8; ++s) async_b128(lv + s * 4096u, Vg, gv + s * 65536);
  };

  stage(0, 0);

  v16bf qa[4];
#pragma unroll
  for (int kc = 0; kc < 4; ++kc) qa[kc] = frag_ld_bf16(Q, DHEAD, row0, kc * 32);

  v8f O[8];
  float m_s[8], l_s[8];
#pragma unroll
  for (int t = 0; t < 8; ++t) { O[t] = (v8f){}; m_s[t] = -INFINITY; l_s[t] = 0.f; }

  for (int jt = 0; jt < SEQ / JTILE; ++jt) {
    int buf = jt & 1;
    if (jt + 1 < SEQ / JTILE) {
      stage((jt + 1) * JTILE, buf ^ 1);  // buf^1 was read last iter; barrier at loop end makes it safe
      wait_async_16();                   // <=16 outstanding -> tile jt landed
      // warm L2 for tile jt+1 (global_prefetch_b8)
      __builtin_prefetch(K + (long)((jt + 1) * JTILE + l * 4) * DHEAD, 0, 1);
    } else {
      wait_async_0();
    }
    __syncthreads();                     // tile jt visible to all waves

    const bf16_t* Kb = Kt + buf * 16384;
    const bf16_t* Vb = Vt + buf * 16384;

    // S = Qp @ K^T for 16 x 128 tile (8 n-subtiles), B frags from LDS
    v8f S[8];
#pragma unroll
    for (int nt = 0; nt < 8; ++nt) {
      v8f acc = {};
#pragma unroll
      for (int kc = 0; kc < 4; ++kc) {
        v16bf b = frag_ld_bf16(Kb, DHEAD, nt * 16, kc * 32);
        acc = wmma_bf16(qa[kc], b, acc);
      }
      S[nt] = acc;
    }

    // online softmax; row M = v + hs*8, stats reduced across the 16-lane half
    float corr[8];
#pragma unroll
    for (int v = 0; v < 8; ++v) {
      float t = S[0][v];
#pragma unroll
      for (int nt = 1; nt < 8; ++nt) t = fmaxf(t, S[nt][v]);
#pragma unroll
      for (int mk = 1; mk <= 8; mk <<= 1) t = fmaxf(t, __shfl_xor(t, mk, 32));
      float mnew = fmaxf(m_s[v], t);
      corr[v] = __expf(m_s[v] - mnew);
      m_s[v] = mnew;
    }
#pragma unroll
    for (int nt = 0; nt < 8; ++nt)
#pragma unroll
      for (int v = 0; v < 8; ++v) S[nt][v] = __expf(S[nt][v] - m_s[v]);
#pragma unroll
    for (int v = 0; v < 8; ++v) {
      float t = 0.f;
#pragma unroll
      for (int nt = 0; nt < 8; ++nt) t += S[nt][v];
#pragma unroll
      for (int mk = 1; mk <= 8; mk <<= 1) t += __shfl_xor(t, mk, 32);
      l_s[v] = l_s[v] * corr[v] + t;
#pragma unroll
      for (int nt = 0; nt < 8; ++nt) O[nt][v] *= corr[v];
    }

    // stage P (C-layout) to per-wave LDS row-major, re-read as A fragments
    bf16_t* Pw = Pst + wave * (16 * JTILE);
#pragma unroll
    for (int nt = 0; nt < 8; ++nt)
#pragma unroll
      for (int v = 0; v < 8; ++v)
        Pw[(v + hs * 8) * JTILE + nt * 16 + r] = (bf16_t)S[nt][v];
    v16bf pa[4];
#pragma unroll
    for (int kc = 0; kc < 4; ++kc) pa[kc] = frag_ld_bf16(Pw, JTILE, 0, kc * 32);

    // O += P @ V ; B fragments are contiguous rows of the LDS V tile ([d][j])
#pragma unroll
    for (int dt = 0; dt < 8; ++dt)
#pragma unroll
      for (int kc = 0; kc < 4; ++kc) {
        v16bf b = frag_ld_bf16(Vb, JTILE, dt * 16, kc * 32);
        O[dt] = wmma_bf16(pa[kc], b, O[dt]);
      }

    __syncthreads();   // all waves done reading buf before it is restaged
  }

  // normalize and store bf16 row-major
#pragma unroll
  for (int dt = 0; dt < 8; ++dt) {
    bf16_t* o = AO + (long)bh * SEQ * DHEAD + (long)(row0 + hs * 8) * DHEAD +
                dt * 16 + r;
#pragma unroll
    for (int v = 0; v < 8; ++v) o[(long)v * DHEAD] = (bf16_t)(O[dt][v] / l_s[v]);
  }
}

// ---------------- Kernel 3: output projection + bias (fp32 out) --------------
__global__ void __launch_bounds__(256)
outproj_kernel(const bf16_t* __restrict__ A, const float* __restrict__ W,
               const float* __restrict__ bias, float* __restrict__ Y) {
  int wave = (int)(threadIdx.x >> 5);
  int row0 = ((int)blockIdx.x * 8 + wave) * 16;
  int l = lane_id(); int r = l & 15; int hs = l >> 4;

  v16bf a[4];
#pragma unroll
  for (int kc = 0; kc < 4; ++kc) a[kc] = frag_ld_bf16(A, DHEAD, row0, kc * 32);

  for (int nt = 0; nt < 8; ++nt) {
    v8f acc = {};
#pragma unroll
    for (int kc = 0; kc < 4; ++kc) {
      v16bf b = frag_ld_f32(W, DHEAD, nt * 16, kc * 32);
      acc = wmma_bf16(a[kc], b, acc);
    }
    float bb = bias[nt * 16 + r];
    float* o = Y + (long)(row0 + hs * 8) * DHEAD + nt * 16 + r;
#pragma unroll
    for (int v = 0; v < 8; ++v) o[(long)v * DHEAD] = acc[v] + bb;
  }
}

// -----------------------------------------------------------------------------
extern "C" void kernel_launch(void* const* d_in, const int* in_sizes, int n_in,
                              void* d_out, int out_size, void* d_ws, size_t ws_size,
                              hipStream_t stream) {
  const float* q    = (const float*)d_in[0];
  const float* kv   = (const float*)d_in[1];
  const float* Wq   = (const float*)d_in[2];
  const float* Wkv  = (const float*)d_in[3];
  const float* Wout = (const float*)d_in[4];
  const float* bout = (const float*)d_in[5];
  float* out = (float*)d_out;

  // workspace: qp | k | vT | attn_out, each 131072*128 bf16 = 32 MB (128 MB total)
  char* ws = (char*)d_ws;
  bf16_t* qp = (bf16_t*)(ws + 0);
  bf16_t* kp = (bf16_t*)(ws + (size_t)33554432);
  bf16_t* vT = (bf16_t*)(ws + (size_t)67108864);
  bf16_t* ao = (bf16_t*)(ws + (size_t)100663296);

  const float SCALE = 0.08838834764831845f;  // 128^-0.5, folded into qp

  proj_kernel<<<NROWS / 128, 256, 0, stream>>>(q, Wq, qp, nullptr, 128, SCALE);
  proj_kernel<<<NROWS / 128, 256, 0, stream>>>(kv, Wkv, kp, vT, 256, 1.0f);
  attn_kernel<<<NBH * 16, 256, 163840, stream>>>(qp, kp, vT, ao);
  outproj_kernel<<<NROWS / 128, 256, 0, stream>>>(ao, Wout, bout, out);
}
